// WaveletDomainProcessing_69561290326634
// MI455X (gfx1250) — matrix operand order repaired and checked by reading, not compiled, not run on hardware
//
#include <hip/hip_runtime.h>
#include <hip/hip_bf16.h>

typedef __bf16 bf16_t;
typedef __attribute__((ext_vector_type(16))) __bf16 v16bf;
typedef __attribute__((ext_vector_type(8)))  float  v8f;
typedef __attribute__((ext_vector_type(4)))  unsigned int u32x4;
typedef __attribute__((ext_vector_type(4)))  int i32x4;
typedef __attribute__((ext_vector_type(8)))  int i32x8;

union BF16x16 { v16bf v; uint4 q[2]; };

#define USE_TDM 1

#define EPSBN 1e-5f
#define BB   8
#define CC   32
#define HH   128
#define WWD  128
#define HW_  (HH * WWD)       // 16384
#define WCH  256              // wavelet channels
#define MIDC 512

// ---------------------------------------------------------------------------
// small utility kernels
// ---------------------------------------------------------------------------
__global__ void zero_f32_k(float* __restrict__ p, int n) {
  int i = blockIdx.x * blockDim.x + threadIdx.x;
  if (i < n) p[i] = 0.0f;
}

__global__ void cvt_f32_bf16_k(const float* __restrict__ in, bf16_t* __restrict__ out, int n) {
  int i = blockIdx.x * blockDim.x + threadIdx.x;
  if (i < n) out[i] = (bf16_t)in[i];
}

// weight transform: OIHW f32 -> [r][cout][cin] bf16  (r = ky*3+kx)
__global__ void wtr_k(const float* __restrict__ w, bf16_t* __restrict__ o,
                      int COUTn, int CINn) {
  long gid = (long)blockIdx.x * blockDim.x + threadIdx.x;
  long total = (long)COUTn * CINn * 9;
  if (gid >= total) return;
  int ci = (int)(gid % CINn);
  long t = gid / CINn;
  int co = (int)(t % COUTn);
  int r  = (int)(t / COUTn);
  o[gid] = (bf16_t)w[((long)co * CINn + ci) * 9 + r];
}

// ---------------------------------------------------------------------------
// grouped directional 5x5 conv: (B,32,H,W) f32 -> (B,256,H,W) bf16, pad=2
// each thread: 4 consecutive x for one (b, oc, y); packed b64 store
// ---------------------------------------------------------------------------
__global__ void dirconv_k(const float* __restrict__ x, const float* __restrict__ wdir,
                          bf16_t* __restrict__ out) {
  long gid = (long)blockIdx.x * blockDim.x + threadIdx.x;
  const long total = (long)BB * WCH * HH * (WWD / 4);
  if (gid >= total) return;
  int x0 = (int)(gid & 31) * 4;          // WWD/4 = 32 quads per row
  int y  = (int)((gid >> 5) & 127);
  int oc = (int)((gid >> 12) & 255);
  int b  = (int)(gid >> 20);
  int ci = oc >> 3;                      // groups = 32, 8 directions per channel
  const float* xin = x + (long)(b * CC + ci) * HW_;
  const float* wf  = wdir + oc * 25;
  float acc[4] = {0.0f, 0.0f, 0.0f, 0.0f};
#pragma unroll
  for (int ky = 0; ky < 5; ++ky) {
    int yy = y + ky - 2;
    if (yy < 0 || yy >= HH) continue;
    float xr[8];
#pragma unroll
    for (int i = 0; i < 8; ++i) {
      int xx = x0 + i - 2;
      xr[i] = ((unsigned)xx < (unsigned)WWD) ? xin[yy * WWD + xx] : 0.0f;
    }
#pragma unroll
    for (int kx = 0; kx < 5; ++kx) {
      float w = wf[ky * 5 + kx];
#pragma unroll
      for (int j = 0; j < 4; ++j) acc[j] += xr[j + kx] * w;
    }
  }
  union { bf16_t h[4]; uint2 q; } pk;
#pragma unroll
  for (int j = 0; j < 4; ++j) pk.h[j] = (bf16_t)acc[j];
  *(uint2*)(out + ((long)(b * WCH + oc) * HH + y) * WWD + x0) = pk.q;
}

// ---------------------------------------------------------------------------
// 3x3 conv as implicit GEMM on v_wmma_f32_16x16x32_bf16, fused bias + BN stats
//   Block tile: 128 couts x 128 pixels (one image row), K-step 32, 8 waves.
//   Wave tile: M=32 x N=64 -> 2 A frags x 4 B frags, 8 accumulators
//   (each B fragment feeds two WMMAs -> 1.5 ds_load_b128 per WMMA).
//   A tile via TDM (tensor_load_to_lds, LDS pad -> pitch 40), B tile via
//   register prefetch + 2x ds_store_b128. Double buffered, 1 barrier / step.
//   Output: bf16 h; per-channel sum/sumsq accumulated to `stats` by atomics.
// ---------------------------------------------------------------------------
template <int CIN, int COUT>
__global__ __launch_bounds__(256)
void conv3x3_wmma_k(const bf16_t* __restrict__ in, const bf16_t* __restrict__ wgt,
                    const float* __restrict__ bias, bf16_t* __restrict__ out,
                    float* __restrict__ stats) {
  constexpr int KC    = CIN / 32;
  constexpr int STEPS = 9 * KC;
  constexpr int AP = 40;              // 80B rows, 16B aligned (TDM pad-compatible)
  constexpr int BP = 40;
  __shared__ __align__(16) bf16_t As[2][128 * AP];   // [cout][k]
  __shared__ __align__(16) bf16_t Bs[2][128 * BP];   // [pixel][k]

  const int tid  = threadIdx.x;
  const int lane = tid & 31;
  const int wave = tid >> 5;
  const int lr   = lane & 15;
  const int hi   = lane >> 4;
  const int wm   = wave >> 1;          // 0..3 : M strip of 32 couts
  const int wn   = wave & 1;           // 0..1 : N strip of 64 pixels

  const int y     = blockIdx.x;        // one image row per block (W == 128)
  const int cout0 = blockIdx.y * 128;
  const int bz    = blockIdx.z;

  v8f acc[2][4];
#pragma unroll
  for (int u = 0; u < 2; ++u)
#pragma unroll
    for (int t = 0; t < 4; ++t) acc[u][t] = v8f{};

  // B staging: thread -> (pixel n, 16 contiguous k)
  const int bn  = tid & 127;
  const int bk0 = (tid >> 7) * 16;
  union { bf16_t b[16]; uint4 q[2]; } bR;

#if !USE_TDM
  uint4 aR0, aR1;
  const int am = tid >> 1;
  const int ak = (tid & 1) * 16;
#endif

  auto stepGeom = [&](int s, int& r9, int& ci0, int& yy, int& kx, bool& rowok) {
    r9  = s / KC;
    ci0 = (s - r9 * KC) * 32;
    int ky = r9 / 3 - 1;
    kx = (r9 - (r9 / 3) * 3) - 1;
    yy = y + ky;
    rowok = (unsigned)yy < (unsigned)HH;
  };

#if USE_TDM
  auto issueTdmA = [&](int s, int buf) {
    int r9, ci0, yy, kx; bool rowok;
    stepGeom(s, r9, ci0, yy, kx, rowok);
    if (wave == 0) {
      unsigned ldsoff = (unsigned)(uintptr_t)(&As[buf][0]);
      unsigned long long ga =
          (unsigned long long)(uintptr_t)(wgt + (long)(r9 * COUT + cout0) * CIN + ci0);
      u32x4 g0;
      g0[0] = 1u;                                    // count=1 valid descriptor
      g0[1] = ldsoff;                                // lds_addr
      g0[2] = (unsigned)(ga & 0xffffffffu);          // global_addr[95:64]
      g0[3] = (unsigned)((ga >> 32) & 0x01ffffffu) | (2u << 30);  // addr hi | type=2
      i32x8 g1;
      // data_size=2B, pad_enable, pad_interval=16DW(code3), pad_amount=4DW(code3)
      g1[0] = (int)((1u << 16) | (1u << 20) | (3u << 22) | (3u << 25));
      g1[1] = (int)(32u << 16);     // tensor_dim0 = 32 (tile == tensor)
      g1[2] = (int)(128u << 16);    // tensor_dim1 = 128
      g1[3] = (int)(32u << 16);     // tile_dim0 = 32
      g1[4] = 128;                  // tile_dim1 = 128
      g1[5] = CIN;                  // tensor_dim0_stride = CIN elements
      g1[6] = 0;
      g1[7] = 0;
      i32x4 gz = {0, 0, 0, 0};
#if defined(__clang_major__) && __clang_major__ >= 23
      i32x8 gz8 = {0, 0, 0, 0, 0, 0, 0, 0};
      __builtin_amdgcn_tensor_load_to_lds(g0, g1, gz, gz, gz8, 0);
#else
      __builtin_amdgcn_tensor_load_to_lds(g0, g1, gz, gz, 0);
#endif
    }
  };
#endif

  auto loadStepB = [&](int s) {
    int r9, ci0, yy, kx; bool rowok;
    stepGeom(s, r9, ci0, yy, kx, rowok);
#if !USE_TDM
    long aoff = (long)(r9 * COUT + cout0 + am) * CIN + ci0 + ak;
    aR0 = *(const uint4*)(wgt + aoff);
    aR1 = *(const uint4*)(wgt + aoff + 8);
#endif
    int xx = bn + kx;
    bool ok = rowok && (unsigned)xx < (unsigned)WWD;
    const bf16_t* src = in + (long)(bz * CIN + ci0 + bk0) * HW_ + yy * WWD + xx;
#pragma unroll
    for (int e = 0; e < 16; ++e) {
      bR.b[e] = ok ? src[(long)e * HW_] : (bf16_t)0.0f;
    }
  };

  auto storeStepB = [&](int buf) {
#if !USE_TDM
    int al = am * AP + ak;
    *(uint4*)&As[buf][al]     = aR0;
    *(uint4*)&As[buf][al + 8] = aR1;
#endif
    bf16_t* dst = &Bs[buf][bn * BP + bk0];
    *(uint4*)dst       = bR.q[0];
    *((uint4*)dst + 1) = bR.q[1];
  };

  // ---- prologue: stage step 0
#if USE_TDM
  issueTdmA(0, 0);
#endif
  loadStepB(0);
  storeStepB(0);
#if USE_TDM
  if (wave == 0) __builtin_amdgcn_s_wait_tensorcnt(0);
#endif
  __syncthreads();

  for (int s = 0; s < STEPS; ++s) {
    const int cur = s & 1;
    const bool more = (s + 1) < STEPS;
    if (more) {
#if USE_TDM
      issueTdmA(s + 1, cur ^ 1);   // DMA next A tile, overlaps WMMAs
#endif
      loadStepB(s + 1);            // global -> regs
    }

    // two A fragments (M strip of 32)
    BF16x16 af[2];
#pragma unroll
    for (int u = 0; u < 2; ++u) {
      const bf16_t* arow = &As[cur][(wm * 32 + u * 16 + lr) * AP];
      af[u].q[0] = *(const uint4*)(arow + hi * 16);
      af[u].q[1] = *(const uint4*)(arow + hi * 16 + 8);
    }
    // 4 B fragments, each feeds both A fragments
#pragma unroll
    for (int t = 0; t < 4; ++t) {
      BF16x16 bf;
      const bf16_t* brow = &Bs[cur][(wn * 64 + t * 16 + lr) * BP + hi * 16];
      bf.q[0] = *(const uint4*)(brow);
      bf.q[1] = *(const uint4*)(brow + 8);
      acc[0][t] = __builtin_amdgcn_wmma_f32_16x16x32_bf16(false, af[0].v, false, bf.v,
                                                          (short)0, acc[0][t], false, false);
      acc[1][t] = __builtin_amdgcn_wmma_f32_16x16x32_bf16(false, af[1].v, false, bf.v,
                                                          (short)0, acc[1][t], false, false);
    }

    if (more) storeStepB(cur ^ 1);
#if USE_TDM
    if (wave == 0 && more) __builtin_amdgcn_s_wait_tensorcnt(0);
#endif
    __syncthreads();
  }

  // ---- epilogue: bias, bf16 store, fused BN partial sums (16-lane reduce)
#pragma unroll
  for (int u = 0; u < 2; ++u) {
#pragma unroll
    for (int v = 0; v < 8; ++v) {
      int co = cout0 + wm * 32 + u * 16 + hi * 8 + v;
      float bv = bias[co];
      long base = (long)(bz * COUT + co) * HW_ + y * WWD + wn * 64;
      float s1 = 0.0f, s2 = 0.0f;
#pragma unroll
      for (int t = 0; t < 4; ++t) {
        float val = acc[u][t][v] + bv;
        out[base + t * 16 + lr] = (bf16_t)val;
        s1 += val;
        s2 += val * val;
      }
#pragma unroll
      for (int msk = 8; msk > 0; msk >>= 1) {
        s1 += __shfl_xor(s1, msk, 16);
        s2 += __shfl_xor(s2, msk, 16);
      }
      if (lr == 0) {
        atomicAdd(&stats[co],        s1);
        atomicAdd(&stats[COUT + co], s2);
      }
    }
  }
}

// ---------------------------------------------------------------------------
// generic small conv (1x1 inverse conv): 32 couts x 128 px tile, f32 out
// ---------------------------------------------------------------------------
template <int CIN, int COUT, int KSZ, int PAD>
__global__ __launch_bounds__(256)
void conv_wmma_k(const bf16_t* __restrict__ in, const bf16_t* __restrict__ wt,
                 const float* __restrict__ bias, float* __restrict__ out) {
  constexpr int KTOT = CIN * KSZ * KSZ;
  constexpr int AP = 40;
  constexpr int BP = 40;
  __shared__ __align__(16) bf16_t As[32 * AP];
  __shared__ __align__(16) bf16_t Bs[128 * BP];

  const int tid  = threadIdx.x;
  const int lane = tid & 31;
  const int wave = tid >> 5;
  const int wm   = wave >> 2;
  const int wn   = wave & 3;
  const int lr   = lane & 15;
  const int hi   = lane >> 4;

  const int pix0  = blockIdx.x * 128;
  const int cout0 = blockIdx.y * 32;
  const int bz    = blockIdx.z;

  v8f acc0 = {};
  v8f acc1 = {};

  for (int k0 = 0; k0 < KTOT; k0 += 32) {
#pragma unroll
    for (int r = 0; r < 4; ++r) {
      int idx = tid + r * 256;
      int m = idx >> 5, k = idx & 31;
      As[m * AP + k] = wt[(long)(cout0 + m) * KTOT + (k0 + k)];
    }
#pragma unroll
    for (int r = 0; r < 16; ++r) {
      int idx = tid + r * 256;
      int k = idx >> 7;
      int n = idx & 127;
      int kg  = k0 + k;
      int ci  = kg / (KSZ * KSZ);
      int rem = kg - ci * (KSZ * KSZ);
      int ky  = rem / KSZ - PAD;
      int kx  = (rem - (rem / KSZ) * KSZ) - PAD;
      int p  = pix0 + n;
      int yy = p / WWD + ky;
      int xx = p % WWD + kx;
      bf16_t v = (bf16_t)0.0f;
      if ((unsigned)yy < (unsigned)HH && (unsigned)xx < (unsigned)WWD)
        v = in[(long)(bz * CIN + ci) * HW_ + yy * WWD + xx];
      Bs[n * BP + k] = v;
    }
    __syncthreads();

    BF16x16 af;
    {
      const bf16_t* arow = &As[(wm * 16 + lr) * AP];
      af.q[0] = *(const uint4*)(arow + hi * 16);
      af.q[1] = *(const uint4*)(arow + hi * 16 + 8);
    }
    BF16x16 bf0, bf1;
    {
      const bf16_t* b0 = &Bs[(wn * 32 + lr) * BP + hi * 16];
      const bf16_t* b1 = &Bs[(wn * 32 + 16 + lr) * BP + hi * 16];
      bf0.q[0] = *(const uint4*)(b0); bf0.q[1] = *(const uint4*)(b0 + 8);
      bf1.q[0] = *(const uint4*)(b1); bf1.q[1] = *(const uint4*)(b1 + 8);
    }
    acc0 = __builtin_amdgcn_wmma_f32_16x16x32_bf16(false, af.v, false, bf0.v,
                                                   (short)0, acc0, false, false);
    acc1 = __builtin_amdgcn_wmma_f32_16x16x32_bf16(false, af.v, false, bf1.v,
                                                   (short)0, acc1, false, false);
    __syncthreads();
  }

#pragma unroll
  for (int v = 0; v < 8; ++v) {
    int m  = hi * 8 + v;
    int co = cout0 + wm * 16 + m;
    float bv = bias[co];
    long base = (long)(bz * COUT + co) * HW_ + pix0 + wn * 32;
    out[base + lr]      = acc0[v] + bv;
    out[base + 16 + lr] = acc1[v] + bv;
  }
}

// ---------------------------------------------------------------------------
// BN finalize (scale/shift from fused stats) and apply (+ReLU, 8-wide bf16)
// ---------------------------------------------------------------------------
__global__ void bn_fin_k(float* __restrict__ stats, const float* __restrict__ g,
                         const float* __restrict__ be, int Cn) {
  int c = blockIdx.x * blockDim.x + threadIdx.x;
  if (c >= Cn) return;
  const float n = (float)((long)BB * HW_);
  float m   = stats[c] / n;
  float var = stats[Cn + c] / n - m * m;
  float sc  = rsqrtf(var + EPSBN) * g[c];
  stats[2 * Cn + c] = sc;
  stats[3 * Cn + c] = be[c] - m * sc;
}

__global__ void bn_apply_k(const bf16_t* __restrict__ h, const float* __restrict__ stats,
                           bf16_t* __restrict__ outn, int Cn) {
  long g8 = (long)blockIdx.x * blockDim.x + threadIdx.x;
  long total8 = (long)BB * Cn * HW_ / 8;
  if (g8 >= total8) return;
  long base = g8 * 8;
  int c = (int)((base >> 14) % Cn);
  float sc = stats[2 * Cn + c];
  float sh = stats[3 * Cn + c];
  union { bf16_t h8[8]; uint4 q; } vi, vo;
  vi.q = *(const uint4*)(h + base);
#pragma unroll
  for (int i = 0; i < 8; ++i) {
    float v = (float)vi.h8[i] * sc + sh;
    vo.h8[i] = (bf16_t)(v > 0.0f ? v : 0.0f);
  }
  *(uint4*)(outn + base) = vo.q;
}

// ---------------------------------------------------------------------------
extern "C" void kernel_launch(void* const* d_in, const int* in_sizes, int n_in,
                              void* d_out, int out_size, void* d_ws, size_t ws_size,
                              hipStream_t stream) {
  const float* x     = (const float*)d_in[0];
  const float* w_dir = (const float*)d_in[1];
  const float* w1    = (const float*)d_in[2];
  const float* b1    = (const float*)d_in[3];
  const float* g1    = (const float*)d_in[4];
  const float* be1   = (const float*)d_in[5];
  const float* w2    = (const float*)d_in[6];
  const float* b2    = (const float*)d_in[7];
  const float* g2    = (const float*)d_in[8];
  const float* be2   = (const float*)d_in[9];
  const float* w_inv = (const float*)d_in[10];
  const float* b_inv = (const float*)d_in[11];
  float* out = (float*)d_out;

  char* ws = (char*)d_ws;
  bf16_t* h_bf = (bf16_t*)(ws + 0);                     // 134.2 MB (h1 bf16, reused h2)
  bf16_t* h1n  = (bf16_t*)(ws + 134217728);             // 134.2 MB
  bf16_t* coef = (bf16_t*)(ws + 268435456);             //  67.1 MB (coeffs, reused h2n)
  bf16_t* w1b  = (bf16_t*)(ws + 335544320);             // [9][512][256] bf16
  bf16_t* w2b  = (bf16_t*)(ws + 337903616);             // [9][256][512] bf16
  bf16_t* wib  = (bf16_t*)(ws + 340262912);             // [32][256] bf16
  float*  st1  = (float*)(ws + 340279296);              // 4*512 floats
  float*  st2  = st1 + 4 * MIDC;                        // 4*256 floats

  // zero BN accumulators every launch (graph replay determinism)
  zero_f32_k<<<(4 * MIDC + 4 * WCH + 255) / 256, 256, 0, stream>>>(st1, 4 * MIDC + 4 * WCH);

  // weight transforms
  wtr_k<<<(MIDC * WCH * 9 + 255) / 256, 256, 0, stream>>>(w1, w1b, MIDC, WCH);
  wtr_k<<<(WCH * MIDC * 9 + 255) / 256, 256, 0, stream>>>(w2, w2b, WCH, MIDC);
  cvt_f32_bf16_k<<<(CC * WCH + 255) / 256, 256, 0, stream>>>(w_inv, wib, CC * WCH);

  // 1) grouped directional 5x5 conv -> bf16 coefficients (4-wide per thread)
  dirconv_k<<<(BB * WCH * HH * (WWD / 4)) / 256, 256, 0, stream>>>(x, w_dir, coef);

  // 2) conv1: 256 -> 512, 3x3 pad1, fused bias + BN stats
  conv3x3_wmma_k<WCH, MIDC><<<dim3(HH, MIDC / 128, BB), 256, 0, stream>>>(
      coef, w1b, b1, h_bf, st1);
  bn_fin_k<<<(MIDC + 255) / 256, 256, 0, stream>>>(st1, g1, be1, MIDC);
  bn_apply_k<<<(BB * MIDC * HW_ / 8 + 255) / 256, 256, 0, stream>>>(h_bf, st1, h1n, MIDC);

  // 3) conv2: 512 -> 256, 3x3 pad1, fused bias + BN stats
  conv3x3_wmma_k<MIDC, WCH><<<dim3(HH, WCH / 128, BB), 256, 0, stream>>>(
      h1n, w2b, b2, h_bf, st2);
  bn_fin_k<<<(WCH + 255) / 256, 256, 0, stream>>>(st2, g2, be2, WCH);
  bn_apply_k<<<(BB * WCH * HW_ / 8 + 255) / 256, 256, 0, stream>>>(h_bf, st2, coef /* h2n */, WCH);

  // 4) inverse 1x1 conv: 256 -> 32 (WMMA) -> final f32 output
  conv_wmma_k<WCH, CC, 1, 0><<<dim3(HW_ / 128, CC / 32, BB), 256, 0, stream>>>(
      coef, wib, b_inv, out);
}
